// MinVarDenseNet_57372173140203
// MI455X (gfx1250) — matrix operand order repaired and checked by reading, not compiled, not run on hardware
//
#include <hip/hip_runtime.h>

// ---------------------------------------------------------------------------
// MinVarDenseNet for MI455X (gfx1250, wave32, WMMA)
//
//   1) bn_cast:   xn = gamma*(x-mean)*rsqrt(var+eps)+beta -> bf16, row stride
//                 padded 10000->10016 (zeros) so GEMM A-staging is unguarded.
//   2) gemm:      Y = relu(xn @ W^T + b) via v_wmma_f32_16x16x32_bf16.
//                 BM=256 (all samples) so the 600 MB W matrix is streamed from
//                 HBM exactly once (23.3 TB/s floor ~26us); xn re-reads stay in
//                 the 192 MB L2. W converted fp32->bf16 while staging into LDS.
//   3) center:    m = Y - mean_rows(Y), emitted i-major (mT) and r-major (mB),
//                 K padded 150->160 with zeros.
//   4) cov_wmma:  A = mT x mB / 149  (== S^T S: eigh-sqrt round trip is the
//                 identity on the PSD Gram matrix).
//   5) fista:     power iteration for lmax + 400 FISTA iterations; the 60-step
//                 bisection projection runs in wave 0 with lane shuffles.
// ---------------------------------------------------------------------------

typedef __attribute__((ext_vector_type(4)))  unsigned int v4u;
typedef __attribute__((ext_vector_type(8)))  unsigned int v8u;
typedef __attribute__((ext_vector_type(8)))  float        v8f;
typedef __attribute__((ext_vector_type(16))) __bf16       v16bf;

#define N_SAMPLES 256
#define N_FEAT    10000
#define KSTRIDE   10016       // N_FEAT padded to multiple of 32
#define N_OUT     15000
#define N_ASSETS  100
#define COV_ROWS  150
#define K_PAD     160         // COV_ROWS padded to multiple of 32
#define BN_EPS    1e-5f
#define TILE_K    32

__device__ __forceinline__ unsigned short f32_to_bf16(float f) {
  __bf16 h = (__bf16)f;                       // hardware cvt (RNE)
  return __builtin_bit_cast(unsigned short, h);
}

__device__ __forceinline__ v16bf make_frag(v4u lo, v4u hi) {
  v8u u;
  u[0] = lo[0]; u[1] = lo[1]; u[2] = lo[2]; u[3] = lo[3];
  u[4] = hi[0]; u[5] = hi[1]; u[6] = hi[2]; u[7] = hi[3];
  return __builtin_bit_cast(v16bf, u);
}

__device__ __forceinline__ v8f wmma_bf16(v16bf a, v16bf b, v8f c) {
  // (neg_a, A, neg_b, B, c_mod, C, reuse_a, reuse_b)
  return __builtin_amdgcn_wmma_f32_16x16x32_bf16(false, a, false, b,
                                                 (short)0, c, false, false);
}

__device__ __forceinline__ float wave_sum(float x) {
#pragma unroll
  for (int m = 16; m; m >>= 1) x += __shfl_xor(x, m, 32);
  return x;
}
__device__ __forceinline__ float wave_min(float x) {
#pragma unroll
  for (int m = 16; m; m >>= 1) x = fminf(x, __shfl_xor(x, m, 32));
  return x;
}
__device__ __forceinline__ float wave_max(float x) {
#pragma unroll
  for (int m = 16; m; m >>= 1) x = fmaxf(x, __shfl_xor(x, m, 32));
  return x;
}
__device__ __forceinline__ float clip1(float x) {
  return fminf(fmaxf(x, -1.0f), 1.0f);
}

// ---------------------------------------------------------------------------
// Kernel 1: BatchNorm (eval) + fp32 -> bf16 cast, K-padded output
// ---------------------------------------------------------------------------
__global__ void __launch_bounds__(256) bn_cast_kernel(
    const float* __restrict__ x, const float* __restrict__ gamma,
    const float* __restrict__ beta, const float* __restrict__ rmean,
    const float* __restrict__ rvar, unsigned short* __restrict__ xn) {
  int idx = blockIdx.x * 256 + threadIdx.x;
  if (idx < N_SAMPLES * KSTRIDE) {
    int f = idx % KSTRIDE;
    int n = idx / KSTRIDE;
    unsigned short v = 0;
    if (f < N_FEAT) {
      float sc = gamma[f] * rsqrtf(rvar[f] + BN_EPS);
      v = f32_to_bf16((x[n * N_FEAT + f] - rmean[f]) * sc + beta[f]);
    }
    xn[idx] = v;
  }
}

// ---------------------------------------------------------------------------
// Kernel 2: Y = relu(xn @ W^T + b), M=256 N=15000 K=10000, bf16 WMMA.
// Block = 256 threads (8 waves). BM=256 x BN=32 tile: W streamed once.
// Wave w: M rows [w*32, w*32+32), both 16-wide N subtiles -> 4 WMMA / K-step.
// ---------------------------------------------------------------------------
#define GEMM_BN 32

__global__ void __launch_bounds__(256) gemm_bias_relu_kernel(
    const unsigned short* __restrict__ Xn,   // (256, KSTRIDE) bf16, zero-pad
    const float* __restrict__ W,             // (15000, 10000) fp32
    const float* __restrict__ bias,          // (15000)
    float* __restrict__ Y) {                 // (256, 15000)
  __shared__ __align__(16) unsigned short As[N_SAMPLES][TILE_K];  // 16 KB
  __shared__ __align__(16) unsigned short Bs[GEMM_BN][TILE_K];    //  2 KB

  const int tid  = threadIdx.x;
  const int lane = tid & 31;
  const int wave = tid >> 5;
  const int gn   = blockIdx.x * GEMM_BN;
  const int kshift = lane >> 4;             // 0: lanes 0-15, 1: lanes 16-31

  // B staging: one float4 of W per thread per K-step
  const int brow = tid >> 3;                // 0..31  (row n within tile)
  const int bc   = (tid & 7) * 4;           // 0..28  (k within tile)
  const int bn   = min(gn + brow, N_OUT - 1);
  const float* wrow = W + (size_t)bn * N_FEAT;

  v8f acc[2][2] = {};

  for (int k0 = 0; k0 < N_FEAT; k0 += TILE_K) {
    // ---- stage A: 256x32 bf16 = 1024 b128 stores, 4 per thread, unguarded
#pragma unroll
    for (int q = 0; q < 4; ++q) {
      int idx = tid + 256 * q;
      int r  = idx >> 2;                    // sample row 0..255
      int c8 = (idx & 3) * 8;               // bf16 col {0,8,16,24}
      *(v4u*)(&As[r][c8]) = *(const v4u*)(Xn + r * KSTRIDE + k0 + c8);
    }
    // ---- stage B: fused fp32 -> bf16 conversion of W ---------------------
    {
      int k = k0 + bc;
      float f0 = 0.0f, f1 = 0.0f, f2 = 0.0f, f3 = 0.0f;
      if (k + 3 < N_FEAT) {                 // uniform except last K-step
        float4 w4 = *(const float4*)(wrow + k);
        f0 = w4.x; f1 = w4.y; f2 = w4.z; f3 = w4.w;
      } else {
        if (k     < N_FEAT) f0 = wrow[k];
        if (k + 1 < N_FEAT) f1 = wrow[k + 1];
        if (k + 2 < N_FEAT) f2 = wrow[k + 2];
        if (k + 3 < N_FEAT) f3 = wrow[k + 3];
      }
      if (k0 + TILE_K < N_FEAT)
        __builtin_prefetch(wrow + k0 + TILE_K + bc, 0, 1);
      unsigned int p0 = ((unsigned int)f32_to_bf16(f1) << 16) | f32_to_bf16(f0);
      unsigned int p1 = ((unsigned int)f32_to_bf16(f3) << 16) | f32_to_bf16(f2);
      *(unsigned long long*)(&Bs[brow][bc]) =
          ((unsigned long long)p1 << 32) | p0;
    }
    __syncthreads();

    // ---- fragments (ISA 7.12.2 layouts) ---------------------------------
    const v4u* ar0 = (const v4u*)(&As[wave * 32 + (lane & 15)][0]);
    const v4u* ar1 = (const v4u*)(&As[wave * 32 + 16 + (lane & 15)][0]);
    v16bf a0 = make_frag(ar0[kshift], ar0[2 + kshift]);
    v16bf a1 = make_frag(ar1[kshift], ar1[2 + kshift]);
    const v4u* br0 = (const v4u*)(&Bs[(lane & 15)][0]);
    const v4u* br1 = (const v4u*)(&Bs[16 + (lane & 15)][0]);
    v16bf b0 = make_frag(br0[kshift], br0[2 + kshift]);
    v16bf b1 = make_frag(br1[kshift], br1[2 + kshift]);

    acc[0][0] = wmma_bf16(a0, b0, acc[0][0]);
    acc[0][1] = wmma_bf16(a0, b1, acc[0][1]);
    acc[1][0] = wmma_bf16(a1, b0, acc[1][0]);
    acc[1][1] = wmma_bf16(a1, b1, acc[1][1]);
    __syncthreads();
  }

  // ---- epilogue: bias + relu --------------------------------------------
#pragma unroll
  for (int mi = 0; mi < 2; ++mi) {
    int mrow0 = wave * 32 + mi * 16 + (kshift << 3);
#pragma unroll
    for (int ni = 0; ni < 2; ++ni) {
      int ncol = gn + ni * 16 + (lane & 15);
      if (ncol < N_OUT) {
        float bv = bias[ncol];
#pragma unroll
        for (int j = 0; j < 8; ++j)
          Y[(mrow0 + j) * N_OUT + ncol] = fmaxf(acc[mi][ni][j] + bv, 0.0f);
      }
    }
  }
}

// ---------------------------------------------------------------------------
// Kernel 3: center rows, write bf16 copies mT (i-major) and mB (r-major)
// ---------------------------------------------------------------------------
__global__ void __launch_bounds__(128) center_kernel(
    const float* __restrict__ Y, unsigned short* __restrict__ mT,
    unsigned short* __restrict__ mB) {
  int s = blockIdx.x;
  int i = threadIdx.x;
  if (i < N_ASSETS) {
    const float* y = Y + s * N_OUT;
    float sum = 0.0f;
    for (int r = 0; r < COV_ROWS; ++r) sum += y[r * N_ASSETS + i];
    float mean = sum * (1.0f / COV_ROWS);
    unsigned short* t = mT + s * (N_ASSETS * K_PAD);
    unsigned short* b = mB + s * (K_PAD * N_ASSETS);
    for (int r = 0; r < COV_ROWS; ++r) {
      unsigned short v = f32_to_bf16(y[r * N_ASSETS + i] - mean);
      t[i * K_PAD + r] = v;
      b[r * N_ASSETS + i] = v;
    }
    for (int r = COV_ROWS; r < K_PAD; ++r) {
      t[i * K_PAD + r] = 0;
      b[r * N_ASSETS + i] = 0;
    }
  }
}

// ---------------------------------------------------------------------------
// Kernel 4: A = (m^T m)/149 per sample via bf16 WMMA. 7x7 C-tiles of 16x16,
// K = 160 (5 steps). One block (4 waves) per sample.
// ---------------------------------------------------------------------------
__global__ void __launch_bounds__(128) cov_wmma_kernel(
    const unsigned short* __restrict__ mT,   // (256,100,160) bf16
    const unsigned short* __restrict__ mB,   // (256,160,100) bf16
    float* __restrict__ Acov) {              // (256,100,100)
  const int s    = blockIdx.x;
  const int tid  = threadIdx.x;
  const int lane = tid & 31;
  const int wave = tid >> 5;
  const int kshift = lane >> 4;

  const unsigned short* at = mT + s * (N_ASSETS * K_PAD);
  const unsigned short* bt = mB + s * (K_PAD * N_ASSETS);
  float* C = Acov + s * (N_ASSETS * N_ASSETS);

  for (int t = wave; t < 49; t += 4) {
    int tm = t / 7, tn = t % 7;
    int i = tm * 16 + (lane & 15); if (i > N_ASSETS - 1) i = N_ASSETS - 1;
    int j = tn * 16 + (lane & 15); if (j > N_ASSETS - 1) j = N_ASSETS - 1;

    v8f acc = {};
    for (int k0 = 0; k0 < K_PAD; k0 += TILE_K) {
      const v4u* ar = (const v4u*)(at + i * K_PAD + k0);
      v16bf afrag = make_frag(ar[kshift], ar[2 + kshift]);

      int kb = k0 + kshift * 8;
      v8u bu;
#pragma unroll
      for (int e = 0; e < 8; ++e) {
        int kk = kb + 2 * e + (e >= 4 ? 8 : 0);
        unsigned int p0 = bt[kk * N_ASSETS + j];
        unsigned int p1 = bt[(kk + 1) * N_ASSETS + j];
        bu[e] = (p1 << 16) | p0;
      }
      v16bf bfrag = __builtin_bit_cast(v16bf, bu);
      acc = wmma_bf16(afrag, bfrag, acc);
    }

    int mrow = tm * 16 + (kshift << 3);
    int ncol = tn * 16 + (lane & 15);
    if (ncol < N_ASSETS) {
#pragma unroll
      for (int jj = 0; jj < 8; ++jj) {
        int r = mrow + jj;
        if (r < N_ASSETS) C[r * N_ASSETS + ncol] = acc[jj] * (1.0f / 149.0f);
      }
    }
  }
}

// ---------------------------------------------------------------------------
// Kernel 5: per-sample FISTA (A in LDS, wave-0 bisection projection)
// ---------------------------------------------------------------------------
__global__ void __launch_bounds__(128) fista_kernel(
    const float* __restrict__ Acov, float* __restrict__ out) {
  __shared__ float As[N_ASSETS * N_ASSETS];
  __shared__ float zs[128];
  __shared__ float vs[128];
  __shared__ float red[4];
  __shared__ float tau_s;

  const int s    = blockIdx.x;
  const int tid  = threadIdx.x;
  const int lane = tid & 31;
  const int wave = tid >> 5;
  const bool rowv = tid < N_ASSETS;

  const float* Ag = Acov + s * (N_ASSETS * N_ASSETS);
  for (int i = tid; i < N_ASSETS * N_ASSETS; i += 128) As[i] = Ag[i];
  __syncthreads();

  // ---- power iteration for lmax ----------------------------------------
  float v = rowv ? 1.0f : 0.0f;
  float lmax = 0.0f;
  for (int it = 0; it < 48; ++it) {
    zs[tid] = rowv ? v : 0.0f;
    __syncthreads();
    float u = 0.0f;
    if (rowv) {
      const float* row = As + tid * N_ASSETS;
      for (int jj = 0; jj < N_ASSETS; ++jj) u += row[jj] * zs[jj];
    }
    float p = wave_sum(u * u);
    if (lane == 0) red[wave] = p;
    __syncthreads();
    lmax = sqrtf(red[0] + red[1] + red[2] + red[3]);
    v = rowv ? u / fmaxf(lmax, 1e-20f) : 0.0f;
    __syncthreads();
  }
  const float step = 1.0f / (2.0f * fmaxf(lmax, 1e-12f));

  // ---- FISTA ------------------------------------------------------------
  float w = rowv ? (1.0f / N_ASSETS) : 0.0f;
  float z = w;
  float t = 1.0f;
  for (int it = 0; it < 400; ++it) {
    zs[tid] = rowv ? z : 0.0f;
    __syncthreads();
    float g = 0.0f;
    if (rowv) {
      const float* row = As + tid * N_ASSETS;
      for (int jj = 0; jj < N_ASSETS; ++jj) g += row[jj] * zs[jj];
    }
    float vi = z - step * 2.0f * g;
    vs[tid] = rowv ? vi : 0.0f;
    __syncthreads();

    if (tid < 32) {                       // wave-0 projection
      float v0 = vs[tid], v1 = vs[tid + 32], v2 = vs[tid + 64];
      bool has3 = (tid + 96) < N_ASSETS;
      float v3 = has3 ? vs[tid + 96] : 0.0f;
      float mn = fminf(fminf(v0, v1), v2); if (has3) mn = fminf(mn, v3);
      float mx = fmaxf(fmaxf(v0, v1), v2); if (has3) mx = fmaxf(mx, v3);
      float lo = wave_min(mn) - 1.0f;
      float hi = wave_max(mx) + 1.0f;
      for (int b = 0; b < 60; ++b) {
        float mid = 0.5f * (lo + hi);
        float c = clip1(v0 - mid) + clip1(v1 - mid) + clip1(v2 - mid)
                  + (has3 ? clip1(v3 - mid) : 0.0f);
        float ssum = wave_sum(c);
        bool big = ssum > 1.0f;
        lo = big ? mid : lo;
        hi = big ? hi : mid;
      }
      float tau0 = 0.5f * (lo + hi);
      float act = (fabsf(v0 - tau0) < 1.0f ? 1.0f : 0.0f)
                + (fabsf(v1 - tau0) < 1.0f ? 1.0f : 0.0f)
                + (fabsf(v2 - tau0) < 1.0f ? 1.0f : 0.0f)
                + ((has3 && fabsf(v3 - tau0) < 1.0f) ? 1.0f : 0.0f);
      float k = fmaxf(wave_sum(act), 1.0f);
      float csum = wave_sum(clip1(v0 - tau0) + clip1(v1 - tau0)
                            + clip1(v2 - tau0)
                            + (has3 ? clip1(v3 - tau0) : 0.0f));
      float tau = tau0 + (csum - 1.0f) / k;
      if (tid == 0) tau_s = tau;
    }
    __syncthreads();
    float tau = tau_s;
    float wn = rowv ? clip1(vi - tau) : 0.0f;
    float tn = 0.5f * (1.0f + sqrtf(1.0f + 4.0f * t * t));
    z = wn + ((t - 1.0f) / tn) * (wn - w);
    w = wn; t = tn;
    __syncthreads();
  }
  if (rowv) out[s * N_ASSETS + tid] = w;
}

// ---------------------------------------------------------------------------
extern "C" void kernel_launch(void* const* d_in, const int* in_sizes, int n_in,
                              void* d_out, int out_size, void* d_ws,
                              size_t ws_size, hipStream_t stream) {
  (void)in_sizes; (void)n_in; (void)out_size; (void)ws_size;
  const float* x     = (const float*)d_in[0];
  const float* gamma = (const float*)d_in[1];
  const float* beta  = (const float*)d_in[2];
  const float* rmean = (const float*)d_in[3];
  const float* rvar  = (const float*)d_in[4];
  const float* W     = (const float*)d_in[5];
  const float* bias  = (const float*)d_in[6];
  float* out = (float*)d_out;

  // Workspace layout (Acov aliases Y, which is dead after center_kernel):
  //   xn  bf16  256*10016*2 =  5,128,192 B
  //   Y   fp32  256*15000*4 = 15,360,000 B   (reused as Acov: 10,240,000 B)
  //   mT  bf16  256*16000*2 =  8,192,000 B
  //   mB  bf16  256*16000*2 =  8,192,000 B
  char* ws = (char*)d_ws;
  unsigned short* xn = (unsigned short*)ws;
  float* Y           = (float*)(ws + 5128192);
  unsigned short* mT = (unsigned short*)(ws + 5128192 + 15360000);
  unsigned short* mB = (unsigned short*)(ws + 5128192 + 15360000 + 8192000);
  float* Acov        = Y;   // Y is consumed by center_kernel before cov writes

  bn_cast_kernel<<<(N_SAMPLES * KSTRIDE + 255) / 256, 256, 0, stream>>>(
      x, gamma, beta, rmean, rvar, xn);

  gemm_bias_relu_kernel<<<(N_OUT + GEMM_BN - 1) / GEMM_BN, 256, 0, stream>>>(
      xn, W, bias, Y);   // 469 blocks

  center_kernel<<<N_SAMPLES, 128, 0, stream>>>(Y, mT, mB);
  cov_wmma_kernel<<<N_SAMPLES, 128, 0, stream>>>(mT, mB, Acov);
  fista_kernel<<<N_SAMPLES, 128, 0, stream>>>(Acov, out);
}
